// Net_54846732370314
// MI455X (gfx1250) — matrix operand order repaired and checked by reading, not compiled, not run on hardware
//
#include <hip/hip_runtime.h>
#include <hip/hip_bf16.h>
#include <math.h>

// ---------------------------------------------------------------------------
// RetinaNet head for MI455X (gfx1250, wave32, WMMA).
// Convs run as implicit GEMM on v_wmma_f32_16x16x32_f16 (f32 accumulate).
// Weights pre-packed to f16 in WMMA A-fragment order; inputs staged per-tap
// in LDS. Each wave computes a 16(cout) x 32(pixel) tile: one A fragment
// feeds two WMMAs, halving weight traffic per output.
// ---------------------------------------------------------------------------

typedef __attribute__((ext_vector_type(16))) _Float16 v16h;
typedef __attribute__((ext_vector_type(8)))  float    v8f;

#define CIN        256
#define KTOT       (CIN * 9)     // 2304
#define NCHUNK     (KTOT / 32)   // 72
#define NUM_CLS    91
#define NUM_ANC    9
#define CAP        4096
#define NEGF       (-1e9f)
#define BBOX_CLIP  4.135166556742356f   // log(1000/16)
#define IMG_W_F    1216.0f
#define IMG_H_F    800.0f
#define LBL_OFF    2017.0f       // IMG_W + IMG_H + 1

// K ordering: k = t*256 + ci  (tap-major; each 32-chunk stays inside one tap)

// ---------------------------------------------------------------------------
// Pack f32 OIHW weights -> f16, WMMA A-fragment order.
// wpack[tile][chunk][lane][e], 16 halfs (32B) contiguous per (tile,chunk,lane).
// A fragment k order used here (contiguous per lane half, not the ISA table order):
//   row = tile*16 + (lane&15); kb = (lane>=16)?16:0
//   klocal = kb + e
// ---------------------------------------------------------------------------
__global__ void pack_weights(const float* __restrict__ wgt,
                             _Float16* __restrict__ wpack,
                             int Cout, int total)
{
    int idx = blockIdx.x * blockDim.x + threadIdx.x;
    if (idx >= total) return;
    int e     = idx & 15;
    int lane  = (idx >> 4) & 31;
    int chunk = (idx >> 9) % NCHUNK;
    int tile  = idx / (NCHUNK * 512);

    int row = tile * 16 + (lane & 15);
    int kb  = (lane >> 4) * 16;
    int klocal = kb + e;
    int k  = chunk * 32 + klocal;
    int t  = k >> 8;       // tap
    int ci = k & 255;

    float v = (row < Cout) ? wgt[(size_t)row * KTOT + ci * 9 + t] : 0.0f;
    wpack[idx] = (_Float16)v;
}

// ---------------------------------------------------------------------------
// Implicit-GEMM 3x3 conv, SAME padding, NCHW. M=Cout, N=pixels, K=2304.
// Pixel tiles are (row y, 32-wide x-range): y is block-uniform -> no per-pixel
// division; scalar row-bounds test per tap; contiguous loads/stores over x.
// One wave = 16(cout) x 32(pixel) tile (2 accumulators, shared A fragment);
// 8 waves/block share the LDS-staged input tap (32 pix x 256 ci as f16).
// ---------------------------------------------------------------------------
__global__ void __launch_bounds__(256)
conv3x3_wmma(const float* __restrict__ in,       // [N, 256, H, W] f32
             const _Float16* __restrict__ wpack, // packed A fragments
             const float* __restrict__ bias,     // [Cout]
             float* __restrict__ out,            // [N, Cout, H, W] f32
             int H, int W, int Cout)
{
    const int HW   = H * W;
    const int tpr  = (W + 31) >> 5;          // 32-wide x-tiles per row
    const int y    = blockIdx.x / tpr;       // block-uniform row
    const int x0   = (blockIdx.x % tpr) << 5;
    const int lane = threadIdx.x & 31;
    const int wave = threadIdx.x >> 5;
    const int n    = blockIdx.z;
    const int coutTile = blockIdx.y * 8 + wave;
    const int coutBase = coutTile * 16;

    // LDS tap stage: [pixel 0..31][ci 0..255], row stride 272 halfs (544B)
    __shared__ __align__(32) _Float16 sB[32][272];

    const float* inN = in + (size_t)n * CIN * HW;

    // staging role: thread = (pix 0..31, klA 0..7); stages 32 contiguous ci
    const int pix = threadIdx.x & 31;
    const int klA = threadIdx.x >> 5;        // 0..7 -> ci block klA*32..+31

    // compute role
    const int col  = lane & 15;              // B/N column within half-tile
    const int klow = (lane < 16) ? 0 : 16;   // B-matrix K base per lane half
    const _Float16* wp = wpack + ((size_t)coutTile * NCHUNK) * 512 + (size_t)lane * 16;

    v8f acc0 = {};
    v8f acc1 = {};

#pragma unroll
    for (int t = 0; t < 9; ++t) {
        const int ty = t / 3 - 1;
        const int tx = t % 3 - 1;
        const int iy = y + ty;
        const int ix = x0 + pix + tx;
        const bool ok = ((unsigned)iy < (unsigned)H) & ((unsigned)ix < (unsigned)W);
        const float* colBase = inN + (long)iy * W + ix;

        __syncthreads();   // previous tap fully consumed
        v16h sv0, sv1;
        if (ok) {
#pragma unroll
            for (int j = 0; j < 16; ++j)
                sv0[j] = (_Float16)colBase[(size_t)(klA * 32 + j) * HW];
#pragma unroll
            for (int j = 0; j < 16; ++j)
                sv1[j] = (_Float16)colBase[(size_t)(klA * 32 + 16 + j) * HW];
        } else {
#pragma unroll
            for (int j = 0; j < 16; ++j) { sv0[j] = (_Float16)0.0f; sv1[j] = (_Float16)0.0f; }
        }
        *(v16h*)&sB[pix][klA * 32]      = sv0;
        *(v16h*)&sB[pix][klA * 32 + 16] = sv1;
        __syncthreads();   // tap staged

#pragma unroll
        for (int c8 = 0; c8 < 8; ++c8) {
            v16h af  = *(const v16h*)(wp + (size_t)(t * 8 + c8) * 512);
            v16h bf0 = *(const v16h*)&sB[col][c8 * 32 + klow];
            v16h bf1 = *(const v16h*)&sB[col + 16][c8 * 32 + klow];
            // D = A x B + C  (neg_a, A, neg_b, B, c_mod, C, reuse_a, reuse_b)
            acc0 = __builtin_amdgcn_wmma_f32_16x16x32_f16(
                false, af, false, bf0, (short)0, acc0, false, false);
            acc1 = __builtin_amdgcn_wmma_f32_16x16x32_f16(
                false, af, false, bf1, (short)0, acc1, false, false);
        }
    }

    // epilogue: C/D layout lane l, vgpr v -> M = v + ((l>=16)?8:0), N = l&15
    const int xs0 = x0 + col;
    const int xs1 = x0 + 16 + col;
    const int rowM = (lane >= 16) ? 8 : 0;
#pragma unroll
    for (int v = 0; v < 8; ++v) {
        int co = coutBase + v + rowM;
        if (co < Cout) {
            float b = bias[co];
            float* orow = out + ((size_t)n * Cout + co) * HW + y * W;
            if (xs0 < W) orow[xs0] = acc0[v] + b;
            if (xs1 < W) orow[xs1] = acc1[v] + b;
        }
    }
}

// ---------------------------------------------------------------------------
// GroupNorm(32 groups over C=256) + ReLU. One block per (n, group).
// ---------------------------------------------------------------------------
__global__ void __launch_bounds__(256)
gn_relu_kernel(const float* __restrict__ in, float* __restrict__ out,
               const float* __restrict__ gamma, const float* __restrict__ beta,
               int H, int W)
{
    const int HW  = H * W;
    const int n   = blockIdx.y;
    const int g   = blockIdx.x;
    const int cpg = 8;
    const int total = cpg * HW;

    const float* base = in  + ((size_t)n * 256 + g * cpg) * HW;
    float*       ob   = out + ((size_t)n * 256 + g * cpg) * HW;

    float s = 0.0f, ss = 0.0f;
    for (int i = threadIdx.x; i < total; i += 256) {
        float v = base[i];
        s += v; ss += v * v;
    }
    __shared__ float rs[256], rq[256];
    rs[threadIdx.x] = s; rq[threadIdx.x] = ss;
    __syncthreads();
    for (int st = 128; st > 0; st >>= 1) {
        if (threadIdx.x < st) {
            rs[threadIdx.x] += rs[threadIdx.x + st];
            rq[threadIdx.x] += rq[threadIdx.x + st];
        }
        __syncthreads();
    }
    const float inv_n = 1.0f / (float)total;
    const float mu  = rs[0] * inv_n;
    const float var = rq[0] * inv_n - mu * mu;
    const float inv = rsqrtf(var + 1e-5f);

    for (int i = threadIdx.x; i < total; i += 256) {
        int   c = g * cpg + i / HW;
        float v = (base[i] - mu) * inv * gamma[c] + beta[c];
        ob[i] = v > 0.0f ? v : 0.0f;
    }
}

// ---------------------------------------------------------------------------
// Sigmoid + threshold, compact surviving (score, flat-index) pairs.
// cls layout [n][819][HW]; reference flat index = (p*9 + a)*91 + c.
// ---------------------------------------------------------------------------
__global__ void score_collect(const float* __restrict__ cls, int HW,
                              float* __restrict__ candS, int* __restrict__ candI,
                              int* __restrict__ count)
{
    const int perImg = NUM_ANC * NUM_CLS * HW;
    int idx = blockIdx.x * blockDim.x + threadIdx.x;
    if (idx >= 2 * perImg) return;
    int n  = idx / perImg;
    int r  = idx % perImg;
    int ch = r / HW;
    int p  = r % HW;
    float sg = 1.0f / (1.0f + expf(-cls[idx]));
    if (sg > 0.05f) {
        int a = ch / NUM_CLS, c = ch % NUM_CLS;
        int pos = atomicAdd(&count[n], 1);
        if (pos < CAP) {
            candS[n * CAP + pos] = sg;
            candI[n * CAP + pos] = (p * NUM_ANC + a) * NUM_CLS + c;
        }
    }
}

// ---------------------------------------------------------------------------
// Iterative top-1000 selection per (image, level). One block per image.
// ---------------------------------------------------------------------------
__global__ void __launch_bounds__(256)
topk_select(const float* __restrict__ candS, const int* __restrict__ candI,
            const int* __restrict__ count,
            float* __restrict__ allS, int* __restrict__ allI, int level)
{
    const int n = blockIdx.x;
    __shared__ float s[CAP];
    __shared__ int   id[CAP];
    __shared__ float rv[256];
    __shared__ int   ri[256];

    int cnt = count[n]; if (cnt > CAP) cnt = CAP;
    for (int i = threadIdx.x; i < CAP; i += 256) {
        s[i]  = (i < cnt) ? candS[n * CAP + i] : -3.4e38f;
        id[i] = (i < cnt) ? candI[n * CAP + i] : 0;
    }
    __syncthreads();

    for (int j = 0; j < 1000; ++j) {
        float best = -3.4e38f; int bi = 0;
        for (int i = threadIdx.x; i < CAP; i += 256)
            if (s[i] > best) { best = s[i]; bi = i; }
        rv[threadIdx.x] = best; ri[threadIdx.x] = bi;
        __syncthreads();
        for (int st = 128; st > 0; st >>= 1) {
            if (threadIdx.x < st && rv[threadIdx.x + st] > rv[threadIdx.x]) {
                rv[threadIdx.x] = rv[threadIdx.x + st];
                ri[threadIdx.x] = ri[threadIdx.x + st];
            }
            __syncthreads();
        }
        if (threadIdx.x == 0) {
            int  o  = n * 5000 + level * 1000 + j;
            float bs = rv[0]; int b = ri[0];
            bool ok = bs > -1e30f;
            allS[o] = ok ? bs : -1.0f;
            allI[o] = ok ? id[b] : 0;
            s[b] = -3.4e38f;
        }
        __syncthreads();
    }
}

// ---------------------------------------------------------------------------
// Box decode + clip for the selected 1000 entries of one level.
// ---------------------------------------------------------------------------
__global__ void decode_kernel(const float* __restrict__ reg,
                              const float* __restrict__ anchors,
                              float* __restrict__ allS, const int* __restrict__ allI,
                              float* __restrict__ allB, int* __restrict__ allL,
                              int HW, int level, int ancOff)
{
    int t = blockIdx.x * blockDim.x + threadIdx.x;
    if (t >= 2000) return;
    int n = t / 1000, j = t % 1000;
    int o = n * 5000 + level * 1000 + j;

    float sc  = allS[o];
    int   idx = allI[o];
    int   ai  = idx / NUM_CLS;
    int   lb  = idx % NUM_CLS;
    int   a   = ai % NUM_ANC;
    int   p   = ai / NUM_ANC;

    const float* anc = anchors + (size_t)(ancOff + ai) * 4;
    float w  = anc[2] - anc[0], h = anc[3] - anc[1];
    float cx = anc[0] + 0.5f * w, cy = anc[1] + 0.5f * h;

    const float* rb = reg + (size_t)n * (NUM_ANC * 4) * HW;
    float dx = rb[(a * 4 + 0) * HW + p];
    float dy = rb[(a * 4 + 1) * HW + p];
    float dw = fminf(rb[(a * 4 + 2) * HW + p], BBOX_CLIP);
    float dh = fminf(rb[(a * 4 + 3) * HW + p], BBOX_CLIP);

    float pcx = dx * w + cx, pcy = dy * h + cy;
    float pw  = expf(dw) * w, ph  = expf(dh) * h;
    float x1 = pcx - 0.5f * pw, y1 = pcy - 0.5f * ph;
    float x2 = pcx + 0.5f * pw, y2 = pcy + 0.5f * ph;
    x1 = fminf(fmaxf(x1, 0.0f), IMG_W_F);
    x2 = fminf(fmaxf(x2, 0.0f), IMG_W_F);
    y1 = fminf(fmaxf(y1, 0.0f), IMG_H_F);
    y2 = fminf(fmaxf(y2, 0.0f), IMG_H_F);

    allB[(size_t)o * 4 + 0] = x1;
    allB[(size_t)o * 4 + 1] = y1;
    allB[(size_t)o * 4 + 2] = x2;
    allB[(size_t)o * 4 + 3] = y2;
    allL[o] = lb;
    allS[o] = (sc > 0.05f) ? sc : NEGF;
}

// ---------------------------------------------------------------------------
// Class-offset NMS over 5000 candidates, 300 picks. One block per image.
// ---------------------------------------------------------------------------
__global__ void __launch_bounds__(1024)
nms_kernel(const float* __restrict__ boxes, const float* __restrict__ scores,
           const int* __restrict__ labels,
           float* __restrict__ outB, float* __restrict__ outS, int* __restrict__ outL)
{
    const int n = blockIdx.x;
    const int tid = threadIdx.x;
    __shared__ float ws[5000];
    __shared__ float rv[1024];
    __shared__ int   ri[1024];

    for (int i = tid; i < 5000; i += 1024) ws[i] = scores[n * 5000 + i];
    __syncthreads();

    for (int it = 0; it < 300; ++it) {
        float best = -3.4e38f; int bi = 0;
        for (int i = tid; i < 5000; i += 1024)
            if (ws[i] > best) { best = ws[i]; bi = i; }
        rv[tid] = best; ri[tid] = bi;
        __syncthreads();
        for (int st = 512; st > 0; st >>= 1) {
            if (tid < st && rv[tid + st] > rv[tid]) {
                rv[tid] = rv[tid + st]; ri[tid] = ri[tid + st];
            }
            __syncthreads();
        }
        int j = ri[0]; float s = rv[0];
        bool valid = s > 0.0f;
        if (tid == 0) {
#pragma unroll
            for (int d = 0; d < 4; ++d)
                outB[(n * 300 + it) * 4 + d] = valid ? boxes[((size_t)n * 5000 + j) * 4 + d] : 0.0f;
            outS[n * 300 + it] = valid ? s : 0.0f;
            outL[n * 300 + it] = valid ? labels[n * 5000 + j] : -1;
        }
        float offj = (float)labels[n * 5000 + j] * LBL_OFF;
        float jx1 = boxes[((size_t)n * 5000 + j) * 4 + 0] + offj;
        float jy1 = boxes[((size_t)n * 5000 + j) * 4 + 1] + offj;
        float jx2 = boxes[((size_t)n * 5000 + j) * 4 + 2] + offj;
        float jy2 = boxes[((size_t)n * 5000 + j) * 4 + 3] + offj;
        float aj  = (jx2 - jx1) * (jy2 - jy1);
        __syncthreads();
        for (int i = tid; i < 5000; i += 1024) {
            float offi = (float)labels[n * 5000 + i] * LBL_OFF;
            float ix1 = boxes[((size_t)n * 5000 + i) * 4 + 0] + offi;
            float iy1 = boxes[((size_t)n * 5000 + i) * 4 + 1] + offi;
            float ix2 = boxes[((size_t)n * 5000 + i) * 4 + 2] + offi;
            float iy2 = boxes[((size_t)n * 5000 + i) * 4 + 3] + offi;
            float lx = fmaxf(jx1, ix1), ly = fmaxf(jy1, iy1);
            float rx = fminf(jx2, ix2), ry = fminf(jy2, iy2);
            float iw = fmaxf(rx - lx, 0.0f), ih = fmaxf(ry - ly, 0.0f);
            float inter = iw * ih;
            float ai2 = (ix2 - ix1) * (iy2 - iy1);
            float iou = inter / (aj + ai2 - inter + 1e-9f);
            if (iou > 0.5f) ws[i] = NEGF;
        }
        if (tid == 0) ws[j] = NEGF;
        __syncthreads();
    }
}

// ---------------------------------------------------------------------------
// Host orchestration
// ---------------------------------------------------------------------------
extern "C" void kernel_launch(void* const* d_in, const int* in_sizes, int n_in,
                              void* d_out, int out_size, void* d_ws, size_t ws_size,
                              hipStream_t stream)
{
    (void)in_sizes; (void)n_in; (void)out_size; (void)ws_size;

    static const int LH[5]     = {100, 50, 25, 13, 7};
    static const int LW[5]     = {152, 76, 38, 19, 10};
    static const int ANCOFF[5] = {0, 136800, 171000, 179550, 181773};
    const int maxHW = 15200;

    const float* cls_tw = (const float*)d_in[5];
    const float* cls_tb = (const float*)d_in[6];
    const float* cls_gs = (const float*)d_in[7];
    const float* cls_gb = (const float*)d_in[8];
    const float* cls_w  = (const float*)d_in[9];
    const float* cls_b  = (const float*)d_in[10];
    const float* reg_tw = (const float*)d_in[11];
    const float* reg_tb = (const float*)d_in[12];
    const float* reg_gs = (const float*)d_in[13];
    const float* reg_gb = (const float*)d_in[14];
    const float* reg_w  = (const float*)d_in[15];
    const float* reg_b  = (const float*)d_in[16];
    const float* anchors = (const float*)d_in[17];

    // Carve workspace
    char* wsp = (char*)d_ws;
    size_t off = 0;
    auto carve = [&](size_t bytes) -> void* {
        void* p = wsp + off;
        off += (bytes + 255) & ~(size_t)255;
        return p;
    };
    float* actA  = (float*)carve((size_t)2 * 256 * maxHW * 4);
    float* actB  = (float*)carve((size_t)2 * 256 * maxHW * 4);
    float* clsO  = (float*)carve((size_t)2 * (NUM_ANC * NUM_CLS) * maxHW * 4);
    float* regO  = (float*)carve((size_t)2 * (NUM_ANC * 4) * maxHW * 4);
    float* candS = (float*)carve((size_t)2 * CAP * 4);
    int*   candI = (int*)  carve((size_t)2 * CAP * 4);
    int*   cnt   = (int*)  carve((size_t)2 * 4);
    float* allS  = (float*)carve((size_t)2 * 5000 * 4);
    int*   allI  = (int*)  carve((size_t)2 * 5000 * 4);
    float* allB  = (float*)carve((size_t)2 * 5000 * 4 * 4);
    int*   allL  = (int*)  carve((size_t)2 * 5000 * 4);

    // packed weights: tiles * 72 chunks * 512 halfs
    const int tilesT = 16;                            // Cout=256
    const int tilesC = (NUM_ANC * NUM_CLS + 15) / 16; // 52
    const int tilesR = (NUM_ANC * 4 + 15) / 16;       // 3
    const size_t packT = (size_t)tilesT * NCHUNK * 512;
    const size_t packC = (size_t)tilesC * NCHUNK * 512;
    const size_t packR = (size_t)tilesR * NCHUNK * 512;
    _Float16* wpClsT = (_Float16*)carve(4 * packT * 2);
    _Float16* wpRegT = (_Float16*)carve(4 * packT * 2);
    _Float16* wpClsH = (_Float16*)carve(packC * 2);
    _Float16* wpRegH = (_Float16*)carve(packR * 2);

    float* outB = (float*)d_out;
    float* outS = outB + 2 * 300 * 4;
    int*   outL = (int*)(outB + 2 * 300 * 4 + 2 * 300);

    const int twStride = 256 * 256 * 9;

    // ---- pack all conv weights once (reused across levels) ----
    for (int l = 0; l < 4; ++l) {
        pack_weights<<<((int)packT + 255) / 256, 256, 0, stream>>>(
            cls_tw + (size_t)l * twStride, wpClsT + l * packT, 256, (int)packT);
        pack_weights<<<((int)packT + 255) / 256, 256, 0, stream>>>(
            reg_tw + (size_t)l * twStride, wpRegT + l * packT, 256, (int)packT);
    }
    pack_weights<<<((int)packC + 255) / 256, 256, 0, stream>>>(
        cls_w, wpClsH, NUM_ANC * NUM_CLS, (int)packC);
    pack_weights<<<((int)packR + 255) / 256, 256, 0, stream>>>(
        reg_w, wpRegH, NUM_ANC * 4, (int)packR);

    for (int lvl = 0; lvl < 5; ++lvl) {
        const int H = LH[lvl], W = LW[lvl], HW = H * W;
        const int TPR = (W + 31) / 32;
        const float* feat = (const float*)d_in[lvl];
        dim3 gridT(H * TPR, (256 + 127) / 128, 2);

        // ---- cls tower + head ----
        const float* x = feat;
        for (int l = 0; l < 4; ++l) {
            conv3x3_wmma<<<gridT, 256, 0, stream>>>(
                x, wpClsT + l * packT, cls_tb + l * 256, actA, H, W, 256);
            gn_relu_kernel<<<dim3(32, 2), 256, 0, stream>>>(
                actA, actB, cls_gs + l * 256, cls_gb + l * 256, H, W);
            x = actB;
        }
        conv3x3_wmma<<<dim3(H * TPR, (NUM_ANC * NUM_CLS + 127) / 128, 2),
                       256, 0, stream>>>(actB, wpClsH, cls_b, clsO, H, W, NUM_ANC * NUM_CLS);

        // ---- reg tower + head ----
        x = feat;
        for (int l = 0; l < 4; ++l) {
            conv3x3_wmma<<<gridT, 256, 0, stream>>>(
                x, wpRegT + l * packT, reg_tb + l * 256, actA, H, W, 256);
            gn_relu_kernel<<<dim3(32, 2), 256, 0, stream>>>(
                actA, actB, reg_gs + l * 256, reg_gb + l * 256, H, W);
            x = actB;
        }
        conv3x3_wmma<<<dim3(H * TPR, (NUM_ANC * 4 + 127) / 128, 2),
                       256, 0, stream>>>(actB, wpRegH, reg_b, regO, H, W, NUM_ANC * 4);

        // ---- per-level postprocess ----
        hipMemsetAsync(cnt, 0, 2 * sizeof(int), stream);
        int tot = 2 * NUM_ANC * NUM_CLS * HW;
        score_collect<<<(tot + 255) / 256, 256, 0, stream>>>(clsO, HW, candS, candI, cnt);
        topk_select<<<2, 256, 0, stream>>>(candS, candI, cnt, allS, allI, lvl);
        decode_kernel<<<(2000 + 255) / 256, 256, 0, stream>>>(
            regO, anchors, allS, allI, allB, allL, HW, lvl, ANCOFF[lvl]);
    }

    nms_kernel<<<2, 1024, 0, stream>>>(allB, allS, allL, outB, outS, outL);
}
